// TemporalHierarchicalTransformer_38843684225996
// MI455X (gfx1250) — compile-verified, matrix operand order checked by reading, not run on hardware
//
#include <hip/hip_runtime.h>
#include <cmath>

// ---------------------------------------------------------------------------
// TemporalHierarchicalTransformer forward for MI455X (gfx1250, wave32).
//
// All matrix math runs on v_wmma_f32_16x16x32_bf16 with fp32 accumulate.
// Producers (embed / LN / attention / GEMM epilogues) dual-write a bf16 copy
// of every tensor that feeds a matrix unit, so the GEMM / attention inner
// loops are pure bf16 b128 loads + WMMA (no fp32->bf16 cvt in steady state,
// half the bytes). Weights are converted to a bf16 arena once per launch.
// Attention is fused flash-style (1 wave per 16-row Q tile) so the 1024x1024
// score matrices are never materialized.
//
// Workspace requirement: ~225 MB (layout in kernel_launch).
// ---------------------------------------------------------------------------

typedef __attribute__((ext_vector_type(16))) __bf16 v16bf;
typedef __attribute__((ext_vector_type(8)))  float  v8f;

constexpr int kDim  = 512;
constexpr int kT    = 1024;
constexpr int kB    = 8;
constexpr int kNH   = 8;
constexpr int kRows = kB * kT;      // 8192
constexpr int kLd   = 3 * kDim;     // 1536 (qkv leading dim, elements)

static __device__ inline v8f zero8() {
  v8f z = {0.f, 0.f, 0.f, 0.f, 0.f, 0.f, 0.f, 0.f};
  return z;
}

// Load a 16-value bf16 fragment for one lane (two 16-byte b128 loads).
// CDNA5 16-bit A/B layout: lane group g (=lane>>4) holds K = k0+g*8+0..7 in
// elements 0..7 and K = k0+16+g*8+0..7 in elements 8..15.
static __device__ inline v16bf load_fragb(const __bf16* __restrict__ rowptr,
                                          int k0, int g) {
  const float4 c0 = *(const float4*)(rowptr + k0 + g * 8);
  const float4 c1 = *(const float4*)(rowptr + k0 + 16 + g * 8);
  v8f tmp;
  tmp[0] = c0.x; tmp[1] = c0.y; tmp[2] = c0.z; tmp[3] = c0.w;
  tmp[4] = c1.x; tmp[5] = c1.y; tmp[6] = c1.z; tmp[7] = c1.w;
  return __builtin_bit_cast(v16bf, tmp);
}

static __device__ inline v8f wmma_bf16(v16bf a, v16bf b, v8f c) {
  // (neg_a, A, neg_b, B, c_mod, C, reuse_a, reuse_b)
  return __builtin_amdgcn_wmma_f32_16x16x32_bf16(false, a, false, b, (short)0, c,
                                                 false, false);
}

static __device__ inline float gelu_f(float x) {
  return 0.5f * x * (1.f + tanhf(0.7978845608028654f * (x + 0.044715f * x * x * x)));
}

// ---------------------------------------------------------------------------
// GEMM: Y[M,N] = X[M,K] * W[N,K]^T + bias ; act 0=none, 1=gelu.
// X, W are bf16; output stored to fp32 (Yf) and/or bf16 (Yb), either nullable.
// Block = 128 threads (4 waves); each wave computes a 32(M) x 64(N) tile:
// per K-step 6 fragment loads feed 8 WMMAs.
// ---------------------------------------------------------------------------
__global__ __launch_bounds__(128) void gemm_kernel(
    const __bf16* __restrict__ Xp, const __bf16* __restrict__ Wp,
    const float* __restrict__ bp, float* __restrict__ Yf,
    __bf16* __restrict__ Yb, int M, int K, int N, int act) {
  const int lane = threadIdx.x & 31;
  const int wave = threadIdx.x >> 5;
  const int r = lane & 15, g = lane >> 4;
  const int m0 = blockIdx.y * 128 + wave * 32;
  const int n0 = blockIdx.x * 64;

  v8f acc0[4], acc1[4];
#pragma unroll
  for (int nt = 0; nt < 4; ++nt) { acc0[nt] = zero8(); acc1[nt] = zero8(); }

  const __bf16* xrow0 = Xp + (size_t)(m0 + r) * K;
  const __bf16* xrow1 = Xp + (size_t)(m0 + 16 + r) * K;
  const __bf16* wrow[4];
#pragma unroll
  for (int nt = 0; nt < 4; ++nt) wrow[nt] = Wp + (size_t)(n0 + nt * 16 + r) * K;

  for (int k0 = 0; k0 < K; k0 += 32) {
    __builtin_prefetch(xrow0 + k0 + 256, 0, 1);  // global_prefetch_b8
    const v16bf a0 = load_fragb(xrow0, k0, g);
    const v16bf a1 = load_fragb(xrow1, k0, g);
#pragma unroll
    for (int nt = 0; nt < 4; ++nt) {
      const v16bf b = load_fragb(wrow[nt], k0, g);
      acc0[nt] = wmma_bf16(a0, b, acc0[nt]);
      acc1[nt] = wmma_bf16(a1, b, acc1[nt]);
    }
  }

#pragma unroll
  for (int nt = 0; nt < 4; ++nt) {
    const int n = n0 + nt * 16 + r;
    const float bv = bp ? bp[n] : 0.f;
#pragma unroll
    for (int i = 0; i < 8; ++i) {
      // C layout: lanes 0-15 -> M = i, lanes 16-31 -> M = 8+i
      const int ma = m0 + g * 8 + i;
      const int mb = ma + 16;
      float va = acc0[nt][i] + bv;
      float vb = acc1[nt][i] + bv;
      if (act == 1) { va = gelu_f(va); vb = gelu_f(vb); }
      if (Yf) { Yf[(size_t)ma * N + n] = va; Yf[(size_t)mb * N + n] = vb; }
      if (Yb) { Yb[(size_t)ma * N + n] = (__bf16)va; Yb[(size_t)mb * N + n] = (__bf16)vb; }
    }
  }
}

// ---------------------------------------------------------------------------
// Fused flash attention. Grid (T/16, NH, B), 1 wave per block.
// qkv: bf16 [B*T, 1536] (q|k|v). out: fp32 [B*T,512] + bf16 copy.
// pm: optional key pad mask (1 = padded -> score -1e9).
// ---------------------------------------------------------------------------
__global__ __launch_bounds__(32) void attn_kernel(
    const __bf16* __restrict__ qkv, float* __restrict__ out,
    __bf16* __restrict__ outb, const int* __restrict__ pm) {
  const int lane = threadIdx.x & 31;
  const int r = lane & 15, g = lane >> 4;
  const int qt = blockIdx.x, h = blockIdx.y, b = blockIdx.z;
  const size_t brow = (size_t)b * kT;

  __shared__ __bf16 plds[16 * 32];  // P tile, C->A layout transpose staging
  __shared__ __bf16 vlds[32 * 64];  // V tile (32 keys x 64 dims)

  const __bf16* qrow = qkv + (brow + qt * 16 + r) * kLd + h * 64;
  const v16bf qa0 = load_fragb(qrow, 0, g);
  const v16bf qa1 = load_fragb(qrow, 32, g);

  v8f acc[4];
#pragma unroll
  for (int dt = 0; dt < 4; ++dt) acc[dt] = zero8();
  float mo[8], l[8];
#pragma unroll
  for (int i = 0; i < 8; ++i) { mo[i] = -3.0e38f; l[i] = 0.f; }

  for (int kb = 0; kb < kT; kb += 32) {
    // Stage V tile: lane owns key row (kb+lane), copies 64 bf16 (128 B).
    {
      const float4* vr =
          (const float4*)(qkv + (brow + kb + lane) * kLd + 1024 + h * 64);
      float4* vl = (float4*)(vlds + lane * 64);
#pragma unroll
      for (int j = 0; j < 8; ++j) vl[j] = vr[j];
    }

    // Scores: S[16 rows x 32 keys] = Q(16x64) . K^T, two n-tiles of 16 keys.
    const __bf16* kr0 = qkv + (brow + kb + r) * kLd + kDim + h * 64;
    const __bf16* kr1 = qkv + (brow + kb + 16 + r) * kLd + kDim + h * 64;
    v8f s0 = zero8(), s1 = zero8();
    s0 = wmma_bf16(qa0, load_fragb(kr0, 0, g), s0);
    s0 = wmma_bf16(qa1, load_fragb(kr0, 32, g), s0);
    s1 = wmma_bf16(qa0, load_fragb(kr1, 0, g), s1);
    s1 = wmma_bf16(qa1, load_fragb(kr1, 32, g), s1);

    bool msk0 = false, msk1 = false;
    if (pm) {
      msk0 = pm[brow + kb + r] != 0;
      msk1 = pm[brow + kb + 16 + r] != 0;
    }

    float p0[8], p1[8], alpha[8];
#pragma unroll
    for (int i = 0; i < 8; ++i) {
      float v0 = msk0 ? -1e9f : s0[i] * 0.125f;  // 1/sqrt(64)
      float v1 = msk1 ? -1e9f : s1[i] * 0.125f;
      float rm = fmaxf(v0, v1);
      rm = fmaxf(rm, __shfl_xor(rm, 1, 32));
      rm = fmaxf(rm, __shfl_xor(rm, 2, 32));
      rm = fmaxf(rm, __shfl_xor(rm, 4, 32));
      rm = fmaxf(rm, __shfl_xor(rm, 8, 32));
      const float mn = fmaxf(mo[i], rm);
      const float al = __expf(mo[i] - mn);
      const float e0 = __expf(v0 - mn);
      const float e1 = __expf(v1 - mn);
      float rs = e0 + e1;
      rs += __shfl_xor(rs, 1, 32);
      rs += __shfl_xor(rs, 2, 32);
      rs += __shfl_xor(rs, 4, 32);
      rs += __shfl_xor(rs, 8, 32);
      l[i] = l[i] * al + rs;
      mo[i] = mn;
      alpha[i] = al;
      p0[i] = e0;
      p1[i] = e1;
    }
#pragma unroll
    for (int dt = 0; dt < 4; ++dt)
#pragma unroll
      for (int i = 0; i < 8; ++i) acc[dt][i] = acc[dt][i] * alpha[i];

    // Stage P (C layout -> row-major LDS, bf16) then reload as an A fragment.
#pragma unroll
    for (int i = 0; i < 8; ++i) {
      const int m = g * 8 + i;
      plds[m * 32 + r] = (__bf16)p0[i];
      plds[m * 32 + 16 + r] = (__bf16)p1[i];
    }
    __syncthreads();
    const v16bf pa = load_fragb(plds + r * 32, 0, g);

    // O(16x64) += P(16x32) . V(32x64), four 16-col d tiles.
#pragma unroll
    for (int dt = 0; dt < 4; ++dt) {
      v16bf vb;
#pragma unroll
      for (int e = 0; e < 16; ++e) {
        const int kk = (e < 8) ? (g * 8 + e) : (16 + g * 8 + (e - 8));
        vb[e] = vlds[kk * 64 + dt * 16 + r];
      }
      acc[dt] = wmma_bf16(pa, vb, acc[dt]);
    }
    __syncthreads();  // protect plds/vlds before next iteration rewrites them
  }

#pragma unroll
  for (int dt = 0; dt < 4; ++dt)
#pragma unroll
    for (int i = 0; i < 8; ++i) {
      const size_t idx =
          (brow + qt * 16 + g * 8 + i) * kDim + h * 64 + dt * 16 + r;
      const float ov = acc[dt][i] / l[i];
      out[idx] = ov;
      outb[idx] = (__bf16)ov;
    }
}

// ---------------------------------------------------------------------------
// LayerNorm over D=512, optional residual, optional ReLU.
// Dual output: fp32 (yf) and bf16 (yb), either nullable.
// ---------------------------------------------------------------------------
__global__ __launch_bounds__(256) void ln_kernel(
    const float* __restrict__ x, const float* __restrict__ res,
    const float* __restrict__ gam, const float* __restrict__ bta,
    float* __restrict__ yf, __bf16* __restrict__ yb, int relu) {
  __shared__ float red[256];
  const int rrow = blockIdx.x, d = threadIdx.x;
  const size_t base = (size_t)rrow << 9;
  float v0 = x[base + d], v1 = x[base + d + 256];
  if (res) { v0 += res[base + d]; v1 += res[base + d + 256]; }
  red[d] = v0 + v1;
  __syncthreads();
  for (int s = 128; s > 0; s >>= 1) {
    if (d < s) red[d] += red[d + s];
    __syncthreads();
  }
  const float mean = red[0] * (1.f / 512.f);
  __syncthreads();
  const float a0 = v0 - mean, a1 = v1 - mean;
  red[d] = a0 * a0 + a1 * a1;
  __syncthreads();
  for (int s = 128; s > 0; s >>= 1) {
    if (d < s) red[d] += red[d + s];
    __syncthreads();
  }
  const float rstd = rsqrtf(red[0] * (1.f / 512.f) + 1e-5f);
  float y0 = a0 * rstd * gam[d] + bta[d];
  float y1 = a1 * rstd * gam[d + 256] + bta[d + 256];
  if (relu) { y0 = fmaxf(y0, 0.f); y1 = fmaxf(y1, 0.f); }
  if (yf) { yf[base + d] = y0; yf[base + d + 256] = y1; }
  if (yb) { yb[base + d] = (__bf16)y0; yb[base + d + 256] = (__bf16)y1; }
}

// ---------------------------------------------------------------------------
__global__ void embed_kernel(const int* __restrict__ tokens,
                             const float* __restrict__ tok_emb,
                             const float* __restrict__ pos_emb,
                             float* __restrict__ outf,
                             __bf16* __restrict__ outb) {
  const int idx = blockIdx.x * blockDim.x + threadIdx.x;  // kRows*512 exact
  const int row = idx >> 9, d = idx & 511;
  const int t = row & (kT - 1);
  const int tok = tokens[row];
  const float v = tok_emb[(size_t)tok * kDim + d] + pos_emb[(size_t)t * kDim + d];
  outf[idx] = v;
  outb[idx] = (__bf16)v;
}

__global__ void cvt_bf16_kernel(const float* __restrict__ s,
                                __bf16* __restrict__ d, int n) {
  const int i = blockIdx.x * blockDim.x + threadIdx.x;
  if (i < n) d[i] = (__bf16)s[i];
}

// strength = sigmoid(dot(k[t-1], q[t]) / 16) > 0.5  <=>  dot > 0
__global__ void bind_dot_kernel(const float* __restrict__ kb,
                                const float* __restrict__ qb,
                                float* __restrict__ bm) {
  const int idx = blockIdx.x * blockDim.x + threadIdx.x;  // kRows exact
  const int t = idx & (kT - 1);
  if (t == 0) { bm[idx] = 0.f; return; }
  const float* kp = kb + (size_t)(idx - 1) * 256;
  const float* qp = qb + (size_t)idx * 256;
  float s = 0.f;
  for (int d = 0; d < 256; ++d) s += kp[d] * qp[d];
  bm[idx] = (s > 0.f) ? 1.f : 0.f;
}

// seg = cumsum(starts) - 1; starts[0]=1, starts[t]=1-bm[t]
__global__ void seg_kernel(const float* __restrict__ bm, int* __restrict__ seg) {
  const int b = threadIdx.x;
  if (b >= kB) return;
  const int base = b * kT;
  int s = 0;
  seg[base] = 0;
  for (int t = 1; t < kT; ++t) {
    if (bm[base + t] <= 0.5f) s++;
    seg[base + t] = s;
  }
}

__global__ void fill_bf16_kernel(__bf16* __restrict__ p, int n) {
  const int i = blockIdx.x * blockDim.x + threadIdx.x;
  if (i < n) p[i] = (__bf16)0.f;
}
__global__ void fill_i32_kernel(int* __restrict__ p, int v, int n) {
  const int i = blockIdx.x * blockDim.x + threadIdx.x;
  if (i < n) p[i] = v;
}

// Segment mean: grid = B, block = 512 (thread per channel). Segments are
// contiguous in t so a running sum suffices. Marks written slots pm=0.
// Mean is consumed only by a GEMM, so it is stored bf16.
__global__ __launch_bounds__(512) void agg_mean_kernel(
    const float* __restrict__ x, const int* __restrict__ seg,
    __bf16* __restrict__ mean, int* __restrict__ pm) {
  const int b = blockIdx.x, d = threadIdx.x;
  const int base = b * kT;
  float s = 0.f;
  int t0 = 0;
  for (int t = 0; t < kT; ++t) {
    s += x[((size_t)(base + t) << 9) + d];
    const int sid = seg[base + t];
    const bool close = (t == kT - 1) || (seg[base + t + 1] != sid);
    if (close) {
      const float cnt = (float)(t - t0 + 1);
      mean[((size_t)(base + sid) << 9) + d] = (__bf16)(s / cnt);
      if (d == 0) pm[base + sid] = 0;
      s = 0.f;
      t0 = t + 1;
    }
  }
}

__global__ void zero_invalid_kernel(float* __restrict__ xf,
                                    __bf16* __restrict__ xb,
                                    const int* __restrict__ pm) {
  const int idx = blockIdx.x * blockDim.x + threadIdx.x;
  if (pm[idx >> 9]) { xf[idx] = 0.f; xb[idx] = (__bf16)0.f; }
}

// out = lo + lo * sigmoid(inf[seg[t]]) ; optional bf16 copy.
__global__ void downward_kernel(const float* __restrict__ lo,
                                const float* __restrict__ inf,
                                const int* __restrict__ seg,
                                float* __restrict__ outf,
                                __bf16* __restrict__ outb) {
  const int idx = blockIdx.x * blockDim.x + threadIdx.x;
  const int row = idx >> 9, d = idx & 511;
  const int b = row >> 10;
  const int s = seg[row];
  const float iv = inf[((size_t)((b << 10) + s) << 9) + d];
  const float lv = lo[idx];
  const float v = lv + lv / (1.f + __expf(-iv));
  outf[idx] = v;
  if (outb) outb[idx] = (__bf16)v;
}

// ---------------------------------------------------------------------------
extern "C" void kernel_launch(void* const* d_in, const int* in_sizes, int n_in,
                              void* d_out, int out_size, void* d_ws,
                              size_t ws_size, hipStream_t stream) {
  (void)in_sizes; (void)n_in; (void)out_size; (void)ws_size;
  // Inputs in setup_inputs() recursive insertion order:
  //  0 tokens, 1 tok_emb, 2 pos_emb,
  //  scales s (base 3+s*12): w_in,b_in,w_out,b_out,ln1_g,ln1_b,
  //                          ff_w1,ff_b1,ff_w2,ff_b2,ln2_g,ln2_b
  //  bind i (base 39+i*4): wk,bk,wq,bq
  //  agg  i (base 47+i*6): w1,b1,ln_g,ln_b,w2,b2
  //  down i (base 59+i*2): w,b            (63 tensors total)
  const int* tokens = (const int*)d_in[0];
  const float* tok_emb = (const float*)d_in[1];
  const float* pos_emb = (const float*)d_in[2];
  auto F = [&](int i) { return (const float*)d_in[i]; };

  // ---- workspace layout: fp32 arena, then int flags, then bf16 arena ----
  const size_t RC = (size_t)kRows * kDim;  // 4,194,304
  float* ws = (float*)d_ws;
  float* S1 = ws;            // attn-out / generic fp32 temp
  float* S2 = S1 + RC;       // proj-out temp
  float* R0 = S2 + RC;
  float* R1 = R0 + RC;
  float* R2 = R1 + RC;
  float* Xb = R2 + RC;       // working activations (embed / aggregated reps)
  float* KQ = Xb + RC;       // binding k | q (8192*256 each)
  float* BM = KQ + RC;       // 8192 binding mask
  int* seg0 = (int*)(BM + kRows);
  int* seg1 = seg0 + kRows;
  int* pm1 = seg1 + kRows;
  int* pm2 = pm1 + kRows;

  __bf16* bfa = (__bf16*)(pm2 + kRows);   // 16B-aligned by construction
  __bf16* WB = bfa;                       // weight cache, 11,534,336 elems
  __bf16* S0b = WB + 11534336;            // 8192*2048 (qkv bf16 / ffn hidden)
  __bf16* S1b = S0b + (size_t)kRows * 2048;  // attn-out bf16
  __bf16* S2b = S1b + RC;                 // segment mean / agg hidden (bf16)
  __bf16* X0b = S2b + RC;                 // bf16 companion of Xb
  __bf16* R0b = X0b + RC;
  __bf16* R1b = R0b + RC;
  __bf16* R2b = R1b + RC;

  const int elemBlocks = (int)(RC / 256);  // 16384

  // ---- one-time (per launch) weight conversion to bf16 arena ----
  size_t woff = 0;
  auto cvtW = [&](int idx, size_t n) -> const __bf16* {
    __bf16* dst = WB + woff;
    woff += n;
    cvt_bf16_kernel<<<(int)((n + 255) / 256), 256, 0, stream>>>(F(idx), dst,
                                                                (int)n);
    return dst;
  };
  const __bf16 *w_in[3], *w_out[3], *ffw1[3], *ffw2[3];
  for (int s = 0; s < 3; ++s) {
    const int pb = 3 + s * 12;
    w_in[s] = cvtW(pb + 0, (size_t)3 * kDim * kDim);
    w_out[s] = cvtW(pb + 2, (size_t)kDim * kDim);
    ffw1[s] = cvtW(pb + 6, (size_t)4 * kDim * kDim);
    ffw2[s] = cvtW(pb + 8, (size_t)4 * kDim * kDim);
  }
  const __bf16 *wkb[2], *wqb[2], *aw1[2], *aw2[2], *dwn[2];
  for (int i = 0; i < 2; ++i) {
    const int bb = 39 + i * 4;
    wkb[i] = cvtW(bb + 0, (size_t)256 * kDim);
    wqb[i] = cvtW(bb + 2, (size_t)256 * kDim);
    const int ab = 47 + i * 6;
    aw1[i] = cvtW(ab + 0, (size_t)kDim * kDim);
    aw2[i] = cvtW(ab + 4, (size_t)kDim * kDim);
    const int db = 59 + i * 2;
    dwn[i] = cvtW(db + 0, (size_t)kDim * kDim);
  }

  auto gemm = [&](const __bf16* Xp, const __bf16* Wp, const float* bp,
                  float* Yf, __bf16* Yb, int M, int K, int N, int act) {
    dim3 grid(N / 64, M / 128);
    gemm_kernel<<<grid, 128, 0, stream>>>(Xp, Wp, bp, Yf, Yb, M, K, N, act);
  };

  auto scale_block = [&](const float* xin, const __bf16* xinb, float* rep,
                         __bf16* repb, int pbase, const int* pmask) {
    gemm(xinb, w_in[(pbase - 3) / 12], F(pbase + 1), nullptr, S0b,
         kRows, kDim, 3 * kDim, 0);                                     // qkv
    attn_kernel<<<dim3(kT / 16, kNH, kB), 32, 0, stream>>>(S0b, S1, S1b, pmask);
    gemm(S1b, w_out[(pbase - 3) / 12], F(pbase + 3), S2, nullptr,
         kRows, kDim, kDim, 0);                                         // proj
    ln_kernel<<<kRows, 256, 0, stream>>>(S2, xin, F(pbase + 4), F(pbase + 5),
                                         rep, repb, 0);
    gemm(repb, ffw1[(pbase - 3) / 12], F(pbase + 7), nullptr, S0b,
         kRows, kDim, 4 * kDim, 1);                                     // ffn1+gelu
    gemm(S0b, ffw2[(pbase - 3) / 12], F(pbase + 9), S1, nullptr,
         kRows, 4 * kDim, kDim, 0);                                     // ffn2
    ln_kernel<<<kRows, 256, 0, stream>>>(S1, rep, F(pbase + 10), F(pbase + 11),
                                         rep, repb, 0);
  };

  auto aggregate = [&](const float* rep, const __bf16* repb, float* dstf,
                       __bf16* dstb, int ii, int* seg, int* pm) {
    const int bb = 39 + ii * 4, ab = 47 + ii * 6;
    float* kbuf = KQ;
    float* qbuf = KQ + (size_t)kRows * 256;
    gemm(repb, wkb[ii], F(bb + 1), kbuf, nullptr, kRows, kDim, 256, 0);
    gemm(repb, wqb[ii], F(bb + 3), qbuf, nullptr, kRows, kDim, 256, 0);
    bind_dot_kernel<<<kRows / 256, 256, 0, stream>>>(kbuf, qbuf, BM);
    seg_kernel<<<1, 32, 0, stream>>>(BM, seg);
    fill_bf16_kernel<<<elemBlocks, 256, 0, stream>>>(S2b, (int)RC);
    fill_i32_kernel<<<kRows / 256, 256, 0, stream>>>(pm, 1, kRows);
    agg_mean_kernel<<<kB, 512, 0, stream>>>(rep, seg, S2b, pm);
    gemm(S2b, aw1[ii], F(ab + 1), S1, nullptr, kRows, kDim, kDim, 0);   // w1
    ln_kernel<<<kRows, 256, 0, stream>>>(S1, nullptr, F(ab + 2), F(ab + 3),
                                         nullptr, S2b, 1);              // ln+relu
    gemm(S2b, aw2[ii], F(ab + 5), dstf, dstb, kRows, kDim, kDim, 0);    // w2
    zero_invalid_kernel<<<elemBlocks, 256, 0, stream>>>(dstf, dstb, pm);
  };

  // h = tok_emb[tokens] + pos_emb
  embed_kernel<<<elemBlocks, 256, 0, stream>>>(tokens, tok_emb, pos_emb, Xb,
                                               X0b);

  scale_block(Xb, X0b, R0, R0b, 3 + 0 * 12, nullptr);
  aggregate(R0, R0b, Xb, X0b, 0, seg0, pm1);
  scale_block(Xb, X0b, R1, R1b, 3 + 1 * 12, pm1);
  aggregate(R1, R1b, Xb, X0b, 1, seg1, pm2);
  scale_block(Xb, X0b, R2, R2b, 3 + 2 * 12, pm2);

  // Downward i=1: reps[1] = down(reps[2], reps[1], seg1); keep bf16 copy.
  gemm(R2b, dwn[1], F(59 + 1 * 2 + 1), S1, nullptr, kRows, kDim, kDim, 0);
  downward_kernel<<<elemBlocks, 256, 0, stream>>>(R1, S1, seg1, R1, R1b);
  // Downward i=0: final output.
  gemm(R1b, dwn[0], F(59 + 0 * 2 + 1), S1, nullptr, kRows, kDim, kDim, 0);
  downward_kernel<<<elemBlocks, 256, 0, stream>>>(R0, S1, seg0, (float*)d_out,
                                                  nullptr);
}
